// GCN_Encoder_29738353557973
// MI455X (gfx1250) — matrix operand order repaired and checked by reading, not compiled, not run on hardware
//
#include <hip/hip_runtime.h>
#include <hip/hip_bf16.h>

typedef __bf16 bf16;
typedef __attribute__((ext_vector_type(16))) __bf16 v16bf;
typedef __attribute__((ext_vector_type(8)))  __bf16 v8bf;
typedef __attribute__((ext_vector_type(8)))  float  v8f;

#define HID 128
#define EPS 1e-5f

// ---------------------------------------------------------------------------
// Utility kernels
// ---------------------------------------------------------------------------
__global__ void k_zero(float* __restrict__ p, long n) {
    long i = (long)blockIdx.x * blockDim.x + threadIdx.x;
    if (i < n) p[i] = 0.0f;
}

__global__ void k_deg_count(const int* __restrict__ src, const int* __restrict__ dst,
                            int E, float* __restrict__ outdeg, float* __restrict__ indeg) {
    int i = blockIdx.x * blockDim.x + threadIdx.x;
    if (i < E) {
        atomicAdd(&outdeg[src[i]], 1.0f);
        atomicAdd(&indeg[dst[i]], 1.0f);
    }
}

__global__ void k_deg_to_scale(float* __restrict__ deg, long n) {
    long i = (long)blockIdx.x * blockDim.x + threadIdx.x;
    if (i < n) deg[i] = rsqrtf(fmaxf(deg[i], 1.0f));
}

// bsum[f] = b[0][f] + b[1][f] + b[2][f]
__global__ void k_bias_sum(const float* __restrict__ b, float* __restrict__ bsum) {
    int f = threadIdx.x;
    bsum[f] = b[f] + b[HID + f] + b[2 * HID + f];
}

// Wt[r][n][k] = (bf16) W[r][k][n]   (transpose + convert; W is [nrel][K][128])
__global__ void k_w_to_bf16t(const float* __restrict__ W, bf16* __restrict__ Wt,
                             int K, long total) {
    long i = (long)blockIdx.x * blockDim.x + threadIdx.x;
    if (i >= total) return;
    int  k  = (int)(i % K);
    long nr = i / K;
    int  n  = (int)(nr % HID);
    long r  = nr / HID;
    Wt[i] = (bf16)W[(r * K + k) * HID + n];
}

// ---------------------------------------------------------------------------
// Edge scatter: agg[dst] += sOut[src]*sIn[dst] * P[src]   (one wave per edge)
// ---------------------------------------------------------------------------
__global__ __launch_bounds__(256)
void k_edge_scatter(const int* __restrict__ src, const int* __restrict__ dst, int E,
                    const float* __restrict__ P, int ldP,
                    const float* __restrict__ sOut, const float* __restrict__ sIn,
                    float* __restrict__ agg) {
    int lane = threadIdx.x & 31;
    long e = (long)blockIdx.x * 8 + (threadIdx.x >> 5);
    if (e >= E) return;
    int s = src[e];
    int d = dst[e];
    float c = sOut[s] * sIn[d];
    const float4 v = *(const float4*)(P + (long)s * ldP + lane * 4);
    float* a = agg + (long)d * HID + lane * 4;
    atomicAdd(a + 0, v.x * c);
    atomicAdd(a + 1, v.y * c);
    atomicAdd(a + 2, v.z * c);
    atomicAdd(a + 3, v.w * c);
}

// ---------------------------------------------------------------------------
// WMMA GEMM:  C[M x (128*gridDim.y)] = (A[M x K] fp32 + abias) @ Bt
//   Bt: pre-transposed bf16 weights, [ntile][128][K] (column-major weight).
//   abias: ALWAYS-VALID fp32 vector of length >= K (zeros when unused).
//   cbias (len 128, optional) + ReLU folded into epilogue.
// Tile: BM=128, BN=128, BK=32. 8 waves (2x4); wave = 64x32 of C
// = 4x2 tiles of v_wmma_f32_16x16x32_bf16.
// ---------------------------------------------------------------------------
__global__ __launch_bounds__(256)
void k_wmma_gemm(const float* __restrict__ A, int lda,
                 const bf16* __restrict__ Btbase,
                 float* __restrict__ C, int ldc,
                 int M, int K,
                 const float* __restrict__ abias,
                 const float* __restrict__ cbias,
                 int relu) {
    __shared__ bf16 As[128 * 32];       // [m][k] row-major, 8 KB
    __shared__ bf16 Bs[128 * 32];       // [n][k], 8 KB

    const int t   = threadIdx.x;
    const int m0  = blockIdx.x * 128;
    const bf16* Bt = Btbase + (size_t)blockIdx.y * HID * K;
    const int cg0 = blockIdx.y * 128;

    // global->LDS assignments (all 32-byte granular)
    const int aRow = t >> 1;            // 0..127
    const int aCol = (t & 1) * 16;      // 0 / 16
    const int bN   = t >> 1;            // 0..127
    const int bKh  = (t & 1) * 16;      // 0 / 16

    const int lane = t & 31;
    const int wid  = t >> 5;
    const int wm   = (wid >> 2) * 64;   // 0 / 64
    const int wn   = (wid & 3) * 32;    // 0..96
    const int lo   = lane & 15;
    const int hi   = lane >> 4;

    // clamped A row pointer (edge tile duplicates last row; discarded at store)
    long grow = m0 + aRow;
    if (grow >= M) grow = M - 1;
    const float* arow = A + grow * (long)lda;

    v8f acc[4][2];
#pragma unroll
    for (int i = 0; i < 4; ++i)
#pragma unroll
        for (int j = 0; j < 2; ++j)
            acc[i][j] = (v8f){0.f, 0.f, 0.f, 0.f, 0.f, 0.f, 0.f, 0.f};

    for (int kt = 0; kt < K; kt += 32) {
        // ---- A tile: 4x float4 global loads, pack to bf16, 2x b128 LDS stores
        {
            const float4* ap = (const float4*)(arow + kt + aCol);
            const float4* bp = (const float4*)(abias + kt + aCol);
            if (kt + 32 < K) __builtin_prefetch((const float*)ap + 32, 0, 1);
            union { bf16 h[16]; uint4 u[2]; } pk;
#pragma unroll
            for (int q = 0; q < 4; ++q) {
                float4 av = ap[q];
                float4 bv = bp[q];
                pk.h[q * 4 + 0] = (bf16)(av.x + bv.x);
                pk.h[q * 4 + 1] = (bf16)(av.y + bv.y);
                pk.h[q * 4 + 2] = (bf16)(av.z + bv.z);
                pk.h[q * 4 + 3] = (bf16)(av.w + bv.w);
            }
            uint4* dst = (uint4*)&As[aRow * 32 + aCol];
            dst[0] = pk.u[0];
            dst[1] = pk.u[1];
        }
        // ---- B tile: already bf16 + transposed; pure b128 copy
        {
            const uint4* sp = (const uint4*)(Bt + (long)bN * K + kt + bKh);
            uint4* dp = (uint4*)&Bs[bN * 32 + bKh];
            dp[0] = sp[0];
            dp[1] = sp[1];
        }
        __syncthreads();

        // ---- fragments per ISA layout
        v16bf afrag[4];
#pragma unroll
        for (int i = 0; i < 4; ++i) {
            const bf16* p = &As[(wm + i * 16 + lo) * 32 + hi * 8];
            v8bf l8 = *(const v8bf*)p;
            v8bf h8 = *(const v8bf*)(p + 16);
            v16bf f;
#pragma unroll
            for (int e = 0; e < 8; ++e) { f[e] = l8[e]; f[e + 8] = h8[e]; }
            afrag[i] = f;
        }
        v16bf bfrag[2];
#pragma unroll
        for (int j = 0; j < 2; ++j) {
            const bf16* p = &Bs[(wn + j * 16 + lo) * 32 + hi * 8];
            v8bf l8 = *(const v8bf*)p;
            v8bf h8 = *(const v8bf*)(p + 16);
            v16bf f;
#pragma unroll
            for (int e = 0; e < 8; ++e) { f[e] = l8[e]; f[e + 8] = h8[e]; }
            bfrag[j] = f;
        }

#pragma unroll
        for (int i = 0; i < 4; ++i)
#pragma unroll
            for (int j = 0; j < 2; ++j)
                acc[i][j] = __builtin_amdgcn_wmma_f32_16x16x32_bf16(
                    false, afrag[i], false, bfrag[j],
                    (short)0, acc[i][j], false, false);

        __syncthreads();
    }

    // ---- epilogue: optional cbias + ReLU, store fp32
#pragma unroll
    for (int i = 0; i < 4; ++i) {
#pragma unroll
        for (int j = 0; j < 2; ++j) {
            const int colL = wn + j * 16 + lo;
            const int col  = cg0 + colL;
            float bb = cbias ? cbias[colL] : 0.0f;
#pragma unroll
            for (int v = 0; v < 8; ++v) {
                long row = m0 + wm + i * 16 + hi * 8 + v;
                if (row < M) {
                    float x = acc[i][j][v] + bb;
                    if (relu) x = fmaxf(x, 0.0f);
                    C[row * (long)ldc + col] = x;
                }
            }
        }
    }
}

// ---------------------------------------------------------------------------
// BatchNorm: per-feature sum / sumsq, then apply
// ---------------------------------------------------------------------------
__global__ __launch_bounds__(128)
void k_bn_stats(const float* __restrict__ x, int M, float* __restrict__ stats) {
    int f = threadIdx.x;                 // 0..127
    long r0 = (long)blockIdx.x * 256;
    float s = 0.f, q = 0.f;
    for (int r = 0; r < 256; ++r) {
        long row = r0 + r;
        if (row < M) {
            float v = x[row * HID + f];
            s += v;
            q += v * v;
        }
    }
    atomicAdd(&stats[f], s);
    atomicAdd(&stats[HID + f], q);
}

__global__ void k_bn_apply(float* __restrict__ x, long total, int M,
                           const float* __restrict__ stats,
                           const float* __restrict__ g,
                           const float* __restrict__ beta) {
    long i = (long)blockIdx.x * blockDim.x + threadIdx.x;
    if (i >= total) return;
    int f = (int)(i & (HID - 1));
    float mean = stats[f] / (float)M;
    float var  = stats[HID + f] / (float)M - mean * mean;
    x[i] = (x[i] - mean) * rsqrtf(var + EPS) * g[f] + beta[f];
}

// ---------------------------------------------------------------------------
// Host orchestration
// ---------------------------------------------------------------------------
static inline int cdiv(long a, long b) { return (int)((a + b - 1) / b); }

extern "C" void kernel_launch(void* const* d_in, const int* in_sizes, int n_in,
                              void* d_out, int out_size, void* d_ws, size_t ws_size,
                              hipStream_t stream) {
    const float* x    = (const float*)d_in[0];
    const int*   seq  = (const int*)d_in[1];
    const int*   knn  = (const int*)d_in[2];
    const int*   dis  = (const int*)d_in[3];
    const float* W0   = (const float*)d_in[4];
    const float* b0   = (const float*)d_in[5];
    const float* fcW0 = (const float*)d_in[6];
    const float* fcb0 = (const float*)d_in[7];
    const float* g0   = (const float*)d_in[8];
    const float* bt0  = (const float*)d_in[9];
    const float* W1   = (const float*)d_in[10];
    const float* b1   = (const float*)d_in[11];
    const float* fcW1 = (const float*)d_in[12];
    const float* fcb1 = (const float*)d_in[13];
    const float* g1   = (const float*)d_in[14];
    const float* bt1  = (const float*)d_in[15];

    const int IN_DIM = 1280;
    const long N  = in_sizes[0] / IN_DIM;
    const int Es = in_sizes[1] / 2, Ek = in_sizes[2] / 2, Ed = in_sizes[3] / 2;
    const int* srcs[3] = {seq, knn, dis};
    const int  Ecnt[3] = {Es, Ek, Ed};

    float* out = (float*)d_out;

    // workspace carve-up (all offsets stay 16B aligned)
    float* scl   = (float*)d_ws;             // 6*N : [sOut0,sIn0,sOut1,sIn1,sOut2,sIn2]
    float* bsum  = scl + 6 * N;              // 128
    float* stats = bsum + HID;               // 256
    float* zbias = stats + 2 * HID;          // 1280 zeros (branch-free "no bias")
    float* P     = zbias + IN_DIM;           // N*384
    float* agg   = P + N * 3 * HID;          // N*128
    float* h0    = agg + N * HID;            // N*128
    bf16* Wt0    = (bf16*)(h0 + N * HID);    // 3*128*1280
    bf16* WtF0   = Wt0 + 3L * HID * IN_DIM;  // 128*128
    bf16* Wt1    = WtF0 + HID * HID;         // 3*128*128
    bf16* WtF1   = Wt1 + 3L * HID * HID;     // 128*128

    const int mt = cdiv(N, 128);             // GEMM row tiles

    // ---- one-time per launch: zero bias, degree scales, weight transposes
    k_zero<<<cdiv(IN_DIM, 256), 256, 0, stream>>>(zbias, IN_DIM);
    k_zero<<<cdiv(6 * N, 256), 256, 0, stream>>>(scl, 6 * N);
    for (int r = 0; r < 3; ++r)
        k_deg_count<<<cdiv(Ecnt[r], 256), 256, 0, stream>>>(
            srcs[r], srcs[r] + Ecnt[r], Ecnt[r], scl + (2 * r) * N, scl + (2 * r + 1) * N);
    k_deg_to_scale<<<cdiv(6 * N, 256), 256, 0, stream>>>(scl, 6 * N);

    {
        long t0 = 3L * HID * IN_DIM;
        k_w_to_bf16t<<<cdiv(t0, 256), 256, 0, stream>>>(W0, Wt0, IN_DIM, t0);
        long t1 = (long)HID * HID;
        k_w_to_bf16t<<<cdiv(t1, 256), 256, 0, stream>>>(fcW0, WtF0, HID, t1);
        long t2 = 3L * HID * HID;
        k_w_to_bf16t<<<cdiv(t2, 256), 256, 0, stream>>>(W1, Wt1, HID, t2);
        k_w_to_bf16t<<<cdiv(t1, 256), 256, 0, stream>>>(fcW1, WtF1, HID, t1);
    }

    // ================= layer 0 =================
    // fused 3-relation projection: P[N,384] = x @ [W0_0|W0_1|W0_2]
    k_wmma_gemm<<<dim3(mt, 3), 256, 0, stream>>>(x, IN_DIM, Wt0, P, 3 * HID,
                                                 (int)N, IN_DIM, zbias, nullptr, 0);
    k_zero<<<cdiv(N * HID, 256), 256, 0, stream>>>(agg, N * HID);
    for (int r = 0; r < 3; ++r)
        k_edge_scatter<<<cdiv(Ecnt[r], 8), 256, 0, stream>>>(
            srcs[r], srcs[r] + Ecnt[r], Ecnt[r], P + r * HID, 3 * HID,
            scl + (2 * r) * N, scl + (2 * r + 1) * N, agg);
    k_bias_sum<<<1, HID, 0, stream>>>(b0, bsum);
    // FC + ReLU (relation bias folded into A, fcb into epilogue)
    k_wmma_gemm<<<dim3(mt, 1), 256, 0, stream>>>(agg, HID, WtF0, h0, HID,
                                                 (int)N, HID, bsum, fcb0, 1);
    k_zero<<<1, 256, 0, stream>>>(stats, 2 * HID);
    k_bn_stats<<<cdiv(N, 256), HID, 0, stream>>>(h0, (int)N, stats);
    k_bn_apply<<<cdiv(N * HID, 256), 256, 0, stream>>>(h0, N * HID, (int)N, stats, g0, bt0);

    // ================= layer 1 =================
    k_wmma_gemm<<<dim3(mt, 3), 256, 0, stream>>>(h0, HID, Wt1, P, 3 * HID,
                                                 (int)N, HID, zbias, nullptr, 0);
    k_zero<<<cdiv(N * HID, 256), 256, 0, stream>>>(agg, N * HID);
    for (int r = 0; r < 3; ++r)
        k_edge_scatter<<<cdiv(Ecnt[r], 8), 256, 0, stream>>>(
            srcs[r], srcs[r] + Ecnt[r], Ecnt[r], P + r * HID, 3 * HID,
            scl + (2 * r) * N, scl + (2 * r + 1) * N, agg);
    k_bias_sum<<<1, HID, 0, stream>>>(b1, bsum);
    k_wmma_gemm<<<dim3(mt, 1), 256, 0, stream>>>(agg, HID, WtF1, out, HID,
                                                 (int)N, HID, bsum, fcb1, 1);
    k_zero<<<1, 256, 0, stream>>>(stats, 2 * HID);
    k_bn_stats<<<cdiv(N, 256), HID, 0, stream>>>(out, (int)N, stats);
    k_bn_apply<<<cdiv(N * HID, 256), 256, 0, stream>>>(out, N * HID, (int)N, stats, g1, bt1);
}
